// SANLayer_88167088652556
// MI455X (gfx1250) — matrix operand order repaired and checked by reading, not compiled, not run on hardware
//
#include <hip/hip_runtime.h>
#include <hip/hip_bf16.h>

// ---------------------------------------------------------------------------
// SAN layer for MI455X (gfx1250, wave32).
//
// Math restructuring (fp32-equivalent to reference):
//   z_branch = alpha @ (H0 + L@(H1 - b1)) + b1     (softmax rows sum to 1)
//   alpha is exactly sparse with L's pattern (exp(-1e20 - max) == 0 in fp32)
//   z_har    = P @ (Hhar - bhar) + bhar
// where Hj = x @ Wjᵀ + bj  (dense GEMMs -> v_wmma_f32_16x16x32_bf16).
// Bulk data movement: the three 67 MB Laplacians are streamed row-by-row into
// LDS with the Tensor Data Mover (tensor_load_to_lds + s_wait_tensorcnt).
// ---------------------------------------------------------------------------

#define NV   4096   // N nodes
#define CH   256    // channels (C_IN == C_OUT)
#define CAP  128    // per-row nnz capacity (density 1%: mean 41, sd 6.4 -> safe)

typedef __attribute__((ext_vector_type(16))) __bf16 v16bf;
typedef __attribute__((ext_vector_type(8)))  float  v8f;
typedef __attribute__((ext_vector_type(4)))  unsigned int u32x4;
typedef __attribute__((ext_vector_type(8)))  int    i32x8;
typedef __attribute__((ext_vector_type(4)))  int    i32x4;

#if __has_builtin(__builtin_amdgcn_tensor_load_to_lds) && __has_builtin(__builtin_amdgcn_s_wait_tensorcnt)
  #define HAVE_TDM 1
  #if __has_include(<hip/amd_detail/amd_gfx1250_TDM.h>)
    #define TDM_6ARG 1   // therock-10.0 headers -> 6-arg builtin
  #endif
#endif

__device__ __forceinline__ __bf16 to_bf(float f) { return (__bf16)f; }

// ---------------------------------------------------------------------------
// H = X @ Wᵀ + bias      X:[NV,CH] row-major, W:[CH,CH] row-major, H:[NV,CH]
// One wave per 16x16 output tile, K-loop fully unrolled (8 x WMMA bf16 k32).
// ---------------------------------------------------------------------------
__global__ __launch_bounds__(128)
void wmma_xwT_kernel(const float* __restrict__ X, const float* __restrict__ W,
                     const float* __restrict__ bias, float* __restrict__ H)
{
    const int lane = threadIdx.x;          // 0..31
    const int half = lane >> 4;            // 0 | 1
    const int lrow = lane & 15;
    const int rowBase = blockIdx.x * 16;
    const int colBase = (blockIdx.y * blockDim.y + threadIdx.y) * 16;

    const int arow = rowBase + lrow;       // A-matrix row held by this lane
    const int bcol = colBase + lrow;       // B-matrix column held by this lane

    v8f acc = {};
    #pragma unroll
    for (int k0 = 0; k0 < CH; k0 += 32) {
        // A fragment (16x32 bf16): lanes<16 hold K = k0+{0..7,16..23},
        // lanes>=16 hold K = k0+{8..15,24..31}  (ISA 16-bit A layout).
        const float* ap = X + (size_t)arow * CH + k0 + half * 8;
        float4 f0 = *(const float4*)(ap + 0);
        float4 f1 = *(const float4*)(ap + 4);
        float4 f2 = *(const float4*)(ap + 16);
        float4 f3 = *(const float4*)(ap + 20);
        v16bf a;
        a[0]  = to_bf(f0.x); a[1]  = to_bf(f0.y); a[2]  = to_bf(f0.z); a[3]  = to_bf(f0.w);
        a[4]  = to_bf(f1.x); a[5]  = to_bf(f1.y); a[6]  = to_bf(f1.z); a[7]  = to_bf(f1.w);
        a[8]  = to_bf(f2.x); a[9]  = to_bf(f2.y); a[10] = to_bf(f2.z); a[11] = to_bf(f2.w);
        a[12] = to_bf(f3.x); a[13] = to_bf(f3.y); a[14] = to_bf(f3.z); a[15] = to_bf(f3.w);

        // B fragment (32x16 bf16): B[k][n] = W[n][k]; lane holds column bcol,
        // same K order as A: lanes<16 K = k0+{0..7,16..23}, lanes>=16 K = k0+{8..15,24..31}.
        const float* bp = W + (size_t)bcol * CH + k0 + half * 8;
        float4 g0 = *(const float4*)(bp + 0);
        float4 g1 = *(const float4*)(bp + 4);
        float4 g2 = *(const float4*)(bp + 16);
        float4 g3 = *(const float4*)(bp + 20);
        v16bf b;
        b[0]  = to_bf(g0.x); b[1]  = to_bf(g0.y); b[2]  = to_bf(g0.z); b[3]  = to_bf(g0.w);
        b[4]  = to_bf(g1.x); b[5]  = to_bf(g1.y); b[6]  = to_bf(g1.z); b[7]  = to_bf(g1.w);
        b[8]  = to_bf(g2.x); b[9]  = to_bf(g2.y); b[10] = to_bf(g2.z); b[11] = to_bf(g2.w);
        b[12] = to_bf(g3.x); b[13] = to_bf(g3.y); b[14] = to_bf(g3.z); b[15] = to_bf(g3.w);

        acc = __builtin_amdgcn_wmma_f32_16x16x32_bf16(
            /*neg_a=*/false, a, /*neg_b=*/false, b,
            /*c_mod=*/(short)0, acc, /*reuse_a=*/false, /*reuse_b=*/false);
    }

    // C/D layout: lane -> column N=lrow, VGPR v -> row M = v + 8*half.
    const float bv = bias[colBase + lrow];
    #pragma unroll
    for (int v = 0; v < 8; ++v) {
        int r = rowBase + v + half * 8;
        H[(size_t)r * CH + colBase + lrow] = acc[v] + bv;
    }
}

// ---------------------------------------------------------------------------
// Attention score vectors: src/dst = x_cat @ att slices, for both branches.
// ---------------------------------------------------------------------------
__global__ __launch_bounds__(256)
void src_dst_kernel(const float* __restrict__ Hi0, const float* __restrict__ Hi1,
                    const float* __restrict__ Hs0, const float* __restrict__ Hs1,
                    const float* __restrict__ att_i, const float* __restrict__ att_s,
                    float* __restrict__ src_i, float* __restrict__ dst_i,
                    float* __restrict__ src_s, float* __restrict__ dst_s)
{
    __shared__ float red[256];
    const int n = blockIdx.x, t = threadIdx.x;
    const float hi0 = Hi0[(size_t)n * CH + t], hi1 = Hi1[(size_t)n * CH + t];
    const float hs0 = Hs0[(size_t)n * CH + t], hs1 = Hs1[(size_t)n * CH + t];
    float v0 = hi0 * att_i[t]       + hi1 * att_i[256 + t];
    float v1 = hi0 * att_i[512 + t] + hi1 * att_i[768 + t];
    float v2 = hs0 * att_s[t]       + hs1 * att_s[256 + t];
    float v3 = hs0 * att_s[512 + t] + hs1 * att_s[768 + t];

    auto reduce = [&](float v) -> float {
        red[t] = v; __syncthreads();
        for (int s = 128; s > 0; s >>= 1) { if (t < s) red[t] += red[t + s]; __syncthreads(); }
        float r = red[0]; __syncthreads();
        return r;
    };
    float r0 = reduce(v0), r1 = reduce(v1), r2 = reduce(v2), r3 = reduce(v3);
    if (t == 0) { src_i[n] = r0; dst_i[n] = r1; src_s[n] = r2; dst_s[n] = r3; }
}

// ---------------------------------------------------------------------------
// Row-wise nonzero compaction of a dense [NV,NV] matrix.
// TDM path: DMA one 16 KB row into LDS (tensor_load_to_lds, TENSORcnt),
// then scan from LDS with b128 reads. Fallback: coalesced float4 global scan.
// ---------------------------------------------------------------------------
__global__ __launch_bounds__(256)
void compact_kernel(const float* __restrict__ L, int* __restrict__ cols,
                    float* __restrict__ vals, int* __restrict__ cnt)
{
    __shared__ int sh_cnt;
    const int n = blockIdx.x, t = threadIdx.x;
    if (t == 0) sh_cnt = 0;

#if defined(HAVE_TDM)
    __shared__ float rowbuf[NV];                       // 16 KB staging tile
    if (t < 32) {  // one wave issues the DMA (TDM issues once per wave, EXEC ignored)
        const unsigned lds_addr = (unsigned)(size_t)rowbuf;   // flat addr[31:0] == LDS byte addr
        const unsigned long long ga =
            (unsigned long long)(size_t)(L + (size_t)n * NV); // tile start in global memory
        // ---- D# group 0: count=1 | lds_addr | global_addr[56:0] | type=2
        u32x4 g0;
        g0[0] = 1u;
        g0[1] = lds_addr;
        g0[2] = (unsigned)(ga & 0xffffffffu);
        g0[3] = (unsigned)((ga >> 32) & 0x01ffffffu) | (2u << 30);
        // ---- D# group 1: data_size=4B; 1-D tile: tile_dim0=NV, tensor_dim0=NV
        i32x8 g1;
        g1[0] = (int)(2u << 16);                 // workgroup_mask=0, data_size=2 (4 bytes)
        g1[1] = (int)((NV & 0xffffu) << 16);     // tensor_dim0[15:0]   (bits 63:48)
        g1[2] = (int)((NV >> 16) & 0xffffu)      // tensor_dim0[31:16]  (bits 79:64)
              | (1 << 16);                       // tensor_dim1 = 1     (bits 111:80)
        g1[3] = (int)((NV & 0xffffu) << 16);     // tile_dim0 = NV      (bits 127:112)
        g1[4] = 0;                               // tile_dim1/2 unused -> 1-D
        g1[5] = NV;                              // tensor_dim0_stride[31:0]
        g1[6] = 0;                               // stride hi / tensor_dim1_stride lo
        g1[7] = 0;
        i32x4 gz = {0, 0, 0, 0};
#if defined(TDM_6ARG)
        i32x8 gz8 = {0, 0, 0, 0, 0, 0, 0, 0};
        __builtin_amdgcn_tensor_load_to_lds(g0, g1, gz, gz, gz8, 0);
#else
        __builtin_amdgcn_tensor_load_to_lds(g0, g1, gz, gz, 0);
#endif
        __builtin_amdgcn_s_wait_tensorcnt(0);
    }
    __syncthreads();
    for (int m = t * 4; m < NV; m += 256 * 4) {
        float4 v = *(const float4*)&rowbuf[m];
        #pragma unroll
        for (int j = 0; j < 4; ++j) {
            float vj = j == 0 ? v.x : j == 1 ? v.y : j == 2 ? v.z : v.w;
            if (vj != 0.f) {
                int slot = atomicAdd(&sh_cnt, 1);
                if (slot < CAP) { cols[n * CAP + slot] = m + j; vals[n * CAP + slot] = vj; }
            }
        }
    }
#else
    __syncthreads();
    const float* row = L + (size_t)n * NV;
    for (int m = t * 4; m < NV; m += 256 * 4) {
        float4 v = *(const float4*)&row[m];
        #pragma unroll
        for (int j = 0; j < 4; ++j) {
            float vj = j == 0 ? v.x : j == 1 ? v.y : j == 2 ? v.z : v.w;
            if (vj != 0.f) {
                int slot = atomicAdd(&sh_cnt, 1);
                if (slot < CAP) { cols[n * CAP + slot] = m + j; vals[n * CAP + slot] = vj; }
            }
        }
    }
#endif
    __syncthreads();
    if (t == 0) cnt[n] = min(sh_cnt, CAP);
}

// ---------------------------------------------------------------------------
// R[n,:] = (Hbase? Hbase[n,:] : 0) + Σ_k v_k*Hmul[col_k,:] - (Σv)*b + (addBias? b : 0)
// (the -(Σv)*b term strips the bias baked into Hmul before the L-product)
// ---------------------------------------------------------------------------
__global__ __launch_bounds__(256)
void spmm_kernel(const int* __restrict__ cols, const float* __restrict__ vals,
                 const int* __restrict__ cnt, const float* __restrict__ Hbase,
                 const float* __restrict__ Hmul, const float* __restrict__ b,
                 float* __restrict__ R, int addBias)
{
    __shared__ int   s_col[CAP];
    __shared__ float s_val[CAP];
    const int n = blockIdx.x, t = threadIdx.x;
    const int c = cnt[n];
    if (t < c) { s_col[t] = cols[n * CAP + t]; s_val[t] = vals[n * CAP + t]; }
    __syncthreads();
    float acc = Hbase ? Hbase[(size_t)n * CH + t] : 0.f;
    float sv = 0.f;
    for (int k = 0; k < c; ++k) {
        if (k + 1 < c)  // hide HBM latency of the random-row gather
            __builtin_prefetch(&Hmul[(size_t)s_col[k + 1] * CH + t], 0, 0);
        acc = fmaf(s_val[k], Hmul[(size_t)s_col[k] * CH + t], acc);
        sv += s_val[k];
    }
    const float bc = b[t];
    acc -= sv * bc;
    if (addBias) acc += bc;
    R[(size_t)n * CH + t] = acc;
}

// ---------------------------------------------------------------------------
// Fused masked softmax + SpMM:
//   e_k = leaky_relu(src[n] + dst[col_k]);  alpha = softmax(e)  (sparse, exact)
//   OUT[n,:] += Σ_k alpha_k * G[col_k,:] + badd        (bias passes softmax)
// ---------------------------------------------------------------------------
__global__ __launch_bounds__(256)
void att_spmm_kernel(const int* __restrict__ cols, const int* __restrict__ cnt,
                     const float* __restrict__ src, const float* __restrict__ dst,
                     const float* __restrict__ G, const float* __restrict__ badd,
                     float* __restrict__ OUT)
{
    __shared__ int   s_col[CAP];
    __shared__ float s_a[CAP];
    __shared__ float red[256];
    const int n = blockIdx.x, t = threadIdx.x;
    const int c = cnt[n];
    const float sn = src[n];

    float e = -3.0e38f;
    if (t < c) {
        int col = cols[n * CAP + t];
        s_col[t] = col;
        float x = sn + dst[col];
        e = x > 0.f ? x : 0.01f * x;      // leaky_relu(0.01)
    }
    red[t] = e; __syncthreads();
    for (int s = 128; s > 0; s >>= 1) { if (t < s) red[t] = fmaxf(red[t], red[t + s]); __syncthreads(); }
    const float mx = red[0]; __syncthreads();

    float w = (t < c) ? expf(e - mx) : 0.f;
    red[t] = w; __syncthreads();
    for (int s = 128; s > 0; s >>= 1) { if (t < s) red[t] += red[t + s]; __syncthreads(); }
    const float inv = 1.f / red[0]; __syncthreads();
    if (t < c) s_a[t] = w * inv;
    __syncthreads();

    float acc = 0.f;
    for (int k = 0; k < c; ++k) {
        if (k + 1 < c)
            __builtin_prefetch(&G[(size_t)s_col[k + 1] * CH + t], 0, 0);
        acc = fmaf(s_a[k], G[(size_t)s_col[k] * CH + t], acc);
    }
    OUT[(size_t)n * CH + t] += acc + badd[t];
}

// ---------------------------------------------------------------------------
extern "C" void kernel_launch(void* const* d_in, const int* in_sizes, int n_in,
                              void* d_out, int out_size, void* d_ws, size_t ws_size,
                              hipStream_t stream)
{
    (void)in_sizes; (void)n_in; (void)out_size; (void)ws_size;
    const float* x      = (const float*)d_in[0];
    const float* Lup    = (const float*)d_in[1];
    const float* Ld     = (const float*)d_in[2];
    const float* P      = (const float*)d_in[3];
    const float* Wirr_w = (const float*)d_in[4];   // [2,256,256]
    const float* Wirr_b = (const float*)d_in[5];   // [2,256]
    const float* Wsol_w = (const float*)d_in[6];
    const float* Wsol_b = (const float*)d_in[7];
    const float* Whar_w = (const float*)d_in[8];   // [256,256]
    const float* Whar_b = (const float*)d_in[9];   // [256]
    const float* att_i  = (const float*)d_in[10];  // [1024]
    const float* att_s  = (const float*)d_in[11];
    float* out = (float*)d_out;

    // Workspace carve-out (~28 MB total)
    char* ws = (char*)d_ws;
    size_t off = 0;
    auto carve = [&](size_t bytes) -> void* {
        void* p = ws + off; off += (bytes + 255) & ~(size_t)255; return p;
    };
    const size_t HB = (size_t)NV * CH * sizeof(float);      // 4 MB each
    float* Hi0  = (float*)carve(HB);
    float* Hi1  = (float*)carve(HB);
    float* Hs0  = (float*)carve(HB);
    float* Hs1  = (float*)carve(HB);
    float* Hh   = (float*)carve(HB);
    float* G    = (float*)carve(HB);
    float* src_i = (float*)carve(NV * sizeof(float));
    float* dst_i = (float*)carve(NV * sizeof(float));
    float* src_s = (float*)carve(NV * sizeof(float));
    float* dst_s = (float*)carve(NV * sizeof(float));
    int*   cols  = (int*)  carve((size_t)NV * CAP * sizeof(int));
    float* vals  = (float*)carve((size_t)NV * CAP * sizeof(float));
    int*   cnt   = (int*)  carve(NV * sizeof(int));

    const dim3 gGrid(NV / 16, CH / 64), gBlk(32, 4);
    // H_j = x @ W_jᵀ + b_j   (5 WMMA GEMMs)
    wmma_xwT_kernel<<<gGrid, gBlk, 0, stream>>>(x, Wirr_w,            Wirr_b,       Hi0);
    wmma_xwT_kernel<<<gGrid, gBlk, 0, stream>>>(x, Wirr_w + CH * CH,  Wirr_b + CH,  Hi1);
    wmma_xwT_kernel<<<gGrid, gBlk, 0, stream>>>(x, Wsol_w,            Wsol_b,       Hs0);
    wmma_xwT_kernel<<<gGrid, gBlk, 0, stream>>>(x, Wsol_w + CH * CH,  Wsol_b + CH,  Hs1);
    wmma_xwT_kernel<<<gGrid, gBlk, 0, stream>>>(x, Whar_w,            Whar_b,       Hh);

    // attention score vectors for both branches
    src_dst_kernel<<<NV, 256, 0, stream>>>(Hi0, Hi1, Hs0, Hs1, att_i, att_s,
                                           src_i, dst_i, src_s, dst_s);

    // harmonic path: OUT = P @ (Hh - bhar) + bhar   (writes OUT)
    compact_kernel<<<NV, 256, 0, stream>>>(P, cols, vals, cnt);
    spmm_kernel<<<NV, 256, 0, stream>>>(cols, vals, cnt, nullptr, Hh, Whar_b, out, 1);

    // irrotational branch (uses Ld)
    compact_kernel<<<NV, 256, 0, stream>>>(Ld, cols, vals, cnt);
    spmm_kernel<<<NV, 256, 0, stream>>>(cols, vals, cnt, Hi0, Hi1, Wirr_b + CH, G, 0);
    att_spmm_kernel<<<NV, 256, 0, stream>>>(cols, cnt, src_i, dst_i, G, Wirr_b + CH, out);

    // solenoidal branch (uses Lup)
    compact_kernel<<<NV, 256, 0, stream>>>(Lup, cols, vals, cnt);
    spmm_kernel<<<NV, 256, 0, stream>>>(cols, vals, cnt, Hs0, Hs1, Wsol_b + CH, G, 0);
    att_spmm_kernel<<<NV, 256, 0, stream>>>(cols, cnt, src_s, dst_s, G, Wsol_b + CH, out);
}
